// MultiHeadAttention_22668837388450
// MI455X (gfx1250) — matrix operand order repaired and checked by reading, compiled, not run on hardware
//
#include <hip/hip_runtime.h>

// ---------------------------------------------------------------------------
// Types for CDNA5 WMMA (wave32): 16x16x32 bf16, f32 accumulate.
// ---------------------------------------------------------------------------
typedef __attribute__((ext_vector_type(16))) __bf16 v16bf;
typedef __attribute__((ext_vector_type(8)))  float  v8f;

union FragU {
    uint4  q[2];
    v16bf  v;
};

__device__ __forceinline__ v16bf make_frag(uint4 lo, uint4 hi) {
    FragU u;
    u.q[0] = lo;
    u.q[1] = hi;
    return u.v;
}

// A-matrix fragment (16x32 bf16, MxK), source row-major with given element stride.
// ISA layout: lane L holds row M=L%16; a[0..7] -> K = kb + half*8 + i,
//             a[8..15] -> K = kb + 16 + half*8 + i   (half = L/16)
__device__ __forceinline__ v16bf load_a_frag(const __bf16* base, int stride,
                                             int m0, int kb, int lane) {
    int half = lane >> 4, ln = lane & 15;
    const __bf16* p = base + (size_t)(m0 + ln) * stride + kb + half * 8;
    return make_frag(*(const uint4*)p, *(const uint4*)(p + 16));
}

// B-matrix fragment (32x16 bf16, KxN), source is "N-major": row n holds K contiguous.
// ISA layout: lane L holds col N=L%16; b[0..7] -> K = kb + half*8 + i, b[8..15] -> K = kb + 16 + half*8 + i
__device__ __forceinline__ v16bf load_b_frag(const __bf16* base, int stride,
                                             int n0, int kb, int lane) {
    int half = lane >> 4, ln = lane & 15;
    const __bf16* p = base + (size_t)(n0 + ln) * stride + kb + half * 8;
    return make_frag(*(const uint4*)p, *(const uint4*)(p + 16));
}

// ---------------------------------------------------------------------------
// CDNA5 async global->LDS copy (16 bytes per lane), ASYNCcnt-tracked.
// Inline asm (portable across ROCm 7.2 / amdgpu-toolchain builtin arity skew).
// ---------------------------------------------------------------------------
__device__ __forceinline__ void async_copy_b128(unsigned lds_off, const void* gptr) {
    asm volatile("global_load_async_to_lds_b128 %0, %1, off"
                 :: "v"(lds_off), "v"(gptr)
                 : "memory");
}

__device__ __forceinline__ void wait_async0() {
    asm volatile("s_wait_asynccnt 0x0" ::: "memory");
}

// ---------------------------------------------------------------------------
// Elementwise f32 -> bf16 convert
// ---------------------------------------------------------------------------
__global__ __launch_bounds__(256) void cvt_bf16_kernel(const float* __restrict__ in,
                                                       __bf16* __restrict__ out, int n) {
    int i = blockIdx.x * 256 + threadIdx.x;
    if (i < n) out[i] = (__bf16)in[i];
}

// ---------------------------------------------------------------------------
// Transpose + convert: W f32 [K,N] row-major -> WT bf16 [N,K] row-major
// ---------------------------------------------------------------------------
__global__ __launch_bounds__(256) void transpose_cvt_kernel(const float* __restrict__ W,
                                                            __bf16* __restrict__ WT, int dim) {
    __shared__ float t[32][33];
    int bx = blockIdx.x * 32, by = blockIdx.y * 32;
    int x = threadIdx.x & 31, y = threadIdx.x >> 5;  // y in 0..7
#pragma unroll
    for (int i = 0; i < 32; i += 8)
        t[y + i][x] = W[(size_t)(by + y + i) * dim + bx + x];
    __syncthreads();
#pragma unroll
    for (int i = 0; i < 32; i += 8)
        WT[(size_t)(bx + y + i) * dim + by + x] = (__bf16)t[x][y + i];
}

// ---------------------------------------------------------------------------
// bf16 GEMM: C[M,N] = A[M,K] * WT[N,K]^T + bias[N]
// Block tile 128x128, BK=64, double-buffered via async global->LDS copies.
// 8 waves (wave32), each wave owns 32x64 (2x4 WMMA tiles), 16 WMMAs per K-tile.
// ---------------------------------------------------------------------------
template <bool OUT_BF16>
__global__ __launch_bounds__(256) void gemm_bf16_kernel(const __bf16* __restrict__ A,
                                                        const __bf16* __restrict__ BT,
                                                        const float* __restrict__ bias,
                                                        void* __restrict__ Cout,
                                                        int M, int N, int K) {
    // 128 rows x 64 K, stride 72 elements (144 B = 16B multiple), x2 buffers.
    __shared__ __align__(16) __bf16 sA[2][128 * 72];
    __shared__ __align__(16) __bf16 sB[2][128 * 72];

    int tid = threadIdx.x;
    int wave = tid >> 5, lane = tid & 31;
    int half = lane >> 4, ln = lane & 15;
    int wr = wave & 3;   // row strip 0..3 -> rows 32*wr .. +31
    int wc = wave >> 2;  // col half 0..1  -> cols 64*wc .. +63
    int m0 = blockIdx.y * 128, n0 = blockIdx.x * 128;

    v8f acc[2][4] = {};

    // Fill mapping: 2 threads per row, 32 elements (4 x b128) each.
    int frow = tid >> 1;          // 0..127
    int fcol = (tid & 1) * 32;    // 0 or 32
    const __bf16* arow = A + (size_t)(m0 + frow) * K + fcol;
    const __bf16* brow = BT + (size_t)(n0 + frow) * K + fcol;
    unsigned aoff0 = (unsigned)(size_t)&sA[0][frow * 72 + fcol];
    unsigned boff0 = (unsigned)(size_t)&sB[0][frow * 72 + fcol];
    const unsigned bufstride = 128u * 72u * 2u;  // bytes per buffer

    auto issue_fill = [&](int buf, int k0) {
        unsigned ao = aoff0 + (unsigned)buf * bufstride;
        unsigned bo = boff0 + (unsigned)buf * bufstride;
#pragma unroll
        for (int s = 0; s < 4; ++s) {
            async_copy_b128(ao + s * 16, arow + k0 + s * 8);
            async_copy_b128(bo + s * 16, brow + k0 + s * 8);
        }
    };

    issue_fill(0, 0);

    int buf = 0;
    for (int k0 = 0; k0 < K; k0 += 64, buf ^= 1) {
        wait_async0();       // own wave's async copies complete
        __syncthreads();     // all waves' copies complete -> tile ready
        if (k0 + 64 < K) issue_fill(buf ^ 1, k0 + 64);

        const __bf16* cA = sA[buf];
        const __bf16* cB = sB[buf];
#pragma unroll
        for (int ks = 0; ks < 2; ++ks) {
            v16bf af[2], bf[4];
#pragma unroll
            for (int i = 0; i < 2; ++i)
                af[i] = load_a_frag(cA, 72, 32 * wr + 16 * i, ks * 32, lane);
#pragma unroll
            for (int j = 0; j < 4; ++j)
                bf[j] = load_b_frag(cB, 72, 64 * wc + 16 * j, ks * 32, lane);
#pragma unroll
            for (int i = 0; i < 2; ++i)
#pragma unroll
                for (int j = 0; j < 4; ++j)
                    acc[i][j] = __builtin_amdgcn_wmma_f32_16x16x32_bf16(
                        false, af[i], false, bf[j], (short)0, acc[i][j], false, false);
        }
        // No trailing barrier: WMMA consumption forces s_wait_dscnt before each
        // wave signals the next iteration's barrier, which precedes the refill
        // of this buffer.
    }

    // Epilogue: C layout (ISA): VGPR r = row r + 8*half, col = lane%16.
#pragma unroll
    for (int i = 0; i < 2; ++i)
#pragma unroll
        for (int j = 0; j < 4; ++j) {
            int col = n0 + 64 * wc + 16 * j + ln;
            float bv = bias[col];
#pragma unroll
            for (int r = 0; r < 8; ++r) {
                int row = m0 + 32 * wr + 16 * i + r + 8 * half;
                float v = acc[i][j][r] + bv;
                if (OUT_BF16)
                    ((__bf16*)Cout)[(size_t)row * N + col] = (__bf16)v;
                else
                    ((float*)Cout)[(size_t)row * N + col] = v;
            }
        }
}

// ---------------------------------------------------------------------------
// Flash-style attention: one block = (batch b, head h, 64-row q block).
// 32 iterations over 64-key tiles. d_k = 64, seq = 2048, D = 1024.
// ---------------------------------------------------------------------------
__global__ __launch_bounds__(256) void attn_kernel(const __bf16* __restrict__ Q,
                                                   const __bf16* __restrict__ Km,
                                                   const __bf16* __restrict__ Vm,
                                                   __bf16* __restrict__ O) {
    const int S = 2048, D = 1024, DK = 64;
    int qb = blockIdx.x;  // 0..31
    int h  = blockIdx.y;  // 0..15
    int b  = blockIdx.z;  // 0..1

    int tid = threadIdx.x;
    int wave = tid >> 5, lane = tid & 31;
    int half = lane >> 4, ln = lane & 15;
    int wr = wave & 3;        // q-row strip: rows 16*wr .. +15 of the 64-row block
    int wc = wave >> 2;       // col half: cols 32*wc .. +31
    int rbase = 16 * wr;
    int cbase = 32 * wc;

    size_t qrow0 = (size_t)b * S + (size_t)qb * 64;

    __shared__ __align__(16) float  sS[64 * 68];     // scores (f32)
    __shared__ __align__(16) __bf16 sP[64 * 72];     // softmax probs (bf16)
    __shared__ __align__(16) __bf16 sVt[64 * 72];    // V tile transposed: [d][kc]
    __shared__ float rowM[64], rowL[64], rowA[64];

    // Preload Q A-fragments (d_k = 64 -> two K-steps of 32), straight from global.
    v16bf qf[2];
    {
        const __bf16* qp = Q + (qrow0 + rbase + ln) * D + h * DK;
#pragma unroll
        for (int ks = 0; ks < 2; ++ks) {
            const __bf16* p = qp + ks * 32 + half * 8;
            qf[ks] = make_frag(*(const uint4*)p, *(const uint4*)(p + 16));
        }
    }

    if (tid < 64) { rowM[tid] = -1e30f; rowL[tid] = 0.0f; }
    v8f oacc[2] = {};
    __syncthreads();

    for (int j = 0; j < 32; ++j) {
        // Prefetch next iteration's K/V tiles into cache (global_prefetch_b8).
        if (j + 1 < 32) {
            size_t nrow = (size_t)b * S + (size_t)(j + 1) * 64 + (tid & 63);
            __builtin_prefetch(Km + nrow * D + h * DK, 0, 3);
            __builtin_prefetch(Vm + nrow * D + h * DK, 0, 3);
        }

        // ---- Stage V tile transposed into LDS: sVt[d][kc] ----
        {
            int kc = tid >> 2;            // 0..63
            int dg = (tid & 3) * 16;      // 0,16,32,48
            const __bf16* vp = Vm + ((size_t)b * S + (size_t)j * 64 + kc) * D + h * DK + dg;
            FragU vv;
            vv.q[0] = *(const uint4*)vp;
            vv.q[1] = *(const uint4*)(vp + 8);
            const __bf16* ve = (const __bf16*)&vv;
#pragma unroll
            for (int e = 0; e < 16; ++e)
                sVt[(dg + e) * 72 + kc] = ve[e];
        }
        __syncthreads();

        // ---- S = Q * K^T (K B-fragments loaded straight from global) ----
        v8f sacc[2] = {};
        const __bf16* kbase = Km + ((size_t)b * S + (size_t)j * 64) * D + h * DK;
#pragma unroll
        for (int t = 0; t < 2; ++t) {
            const __bf16* kp = kbase + (size_t)(cbase + 16 * t + ln) * D;
#pragma unroll
            for (int ks = 0; ks < 2; ++ks) {
                const __bf16* p = kp + ks * 32 + half * 8;
                v16bf kf = make_frag(*(const uint4*)p, *(const uint4*)(p + 16));
                sacc[t] = __builtin_amdgcn_wmma_f32_16x16x32_bf16(
                    false, qf[ks], false, kf, (short)0, sacc[t], false, false);
            }
        }
        // Write scaled scores to LDS.
#pragma unroll
        for (int t = 0; t < 2; ++t) {
            int col = cbase + 16 * t + ln;
#pragma unroll
            for (int r = 0; r < 8; ++r)
                sS[(rbase + r + 8 * half) * 68 + col] = sacc[t][r] * 0.125f;
        }
        __syncthreads();

        // ---- Online softmax (one thread per row) ----
        if (tid < 64) {
            float mo = rowM[tid], m = mo;
            for (int c = 0; c < 64; ++c) m = fmaxf(m, sS[tid * 68 + c]);
            float a = __expf(mo - m);
            float l = 0.0f;
            for (int c = 0; c < 64; ++c) {
                float p = __expf(sS[tid * 68 + c] - m);
                sP[tid * 72 + c] = (__bf16)p;
                l += p;
            }
            rowM[tid] = m;
            rowL[tid] = rowL[tid] * a + l;
            rowA[tid] = a;
        }
        __syncthreads();

        // ---- Rescale O accumulators, then O += P * V ----
#pragma unroll
        for (int r = 0; r < 8; ++r) {
            float a = rowA[rbase + r + 8 * half];
            oacc[0][r] *= a;
            oacc[1][r] *= a;
        }
#pragma unroll
        for (int ks = 0; ks < 2; ++ks) {
            v16bf pf = load_a_frag(sP, 72, rbase, ks * 32, lane);
#pragma unroll
            for (int t = 0; t < 2; ++t) {
                v16bf vf = load_b_frag(sVt, 72, cbase + 16 * t, ks * 32, lane);
                oacc[t] = __builtin_amdgcn_wmma_f32_16x16x32_bf16(
                    false, pf, false, vf, (short)0, oacc[t], false, false);
            }
        }
        __syncthreads();
    }

    // ---- Final 1/l scale and store (bf16, head-interleaved back to [B,S,D]) ----
#pragma unroll
    for (int r = 0; r < 8; ++r) {
        int row = rbase + r + 8 * half;
        float inv = 1.0f / rowL[row];
#pragma unroll
        for (int t = 0; t < 2; ++t) {
            int col = cbase + 16 * t + ln;
            O[(qrow0 + row) * D + h * DK + col] = (__bf16)(oacc[t][r] * inv);
        }
    }
}

// ---------------------------------------------------------------------------
// Host launcher
// ---------------------------------------------------------------------------
extern "C" void kernel_launch(void* const* d_in, const int* in_sizes, int n_in,
                              void* d_out, int out_size, void* d_ws, size_t ws_size,
                              hipStream_t stream) {
    const float* x  = (const float*)d_in[0];
    const float* Wq = (const float*)d_in[1];
    const float* bq = (const float*)d_in[2];
    const float* Wk = (const float*)d_in[3];
    const float* bk = (const float*)d_in[4];
    const float* Wv = (const float*)d_in[5];
    const float* bv = (const float*)d_in[6];
    const float* Wo = (const float*)d_in[7];
    const float* bo = (const float*)d_in[8];

    const int Mrows = 4096;  // B*S
    const int D = 1024;

    char* ws = (char*)d_ws;
    const size_t MB = (size_t)1 << 20;
    __bf16* xb  = (__bf16*)(ws + 0 * MB);   //  8 MB  x in bf16
    __bf16* WqT = (__bf16*)(ws + 8 * MB);   //  2 MB  Wq^T bf16
    __bf16* WkT = (__bf16*)(ws + 10 * MB);
    __bf16* WvT = (__bf16*)(ws + 12 * MB);
    __bf16* WoT = (__bf16*)(ws + 14 * MB);
    __bf16* Qb  = (__bf16*)(ws + 16 * MB);  //  8 MB each
    __bf16* Kb  = (__bf16*)(ws + 24 * MB);
    __bf16* Vb  = (__bf16*)(ws + 32 * MB);
    __bf16* Ab  = (__bf16*)(ws + 40 * MB);  // attention output, bf16

    // 1) Convert x to bf16; transpose+convert the four weight matrices.
    cvt_bf16_kernel<<<(Mrows * D) / 256, 256, 0, stream>>>(x, xb, Mrows * D);
    dim3 tg(D / 32, D / 32);
    transpose_cvt_kernel<<<tg, 256, 0, stream>>>(Wq, WqT, D);
    transpose_cvt_kernel<<<tg, 256, 0, stream>>>(Wk, WkT, D);
    transpose_cvt_kernel<<<tg, 256, 0, stream>>>(Wv, WvT, D);
    transpose_cvt_kernel<<<tg, 256, 0, stream>>>(Wo, WoT, D);

    // 2) Q/K/V projections (bf16 WMMA, f32 accumulate, bias add, bf16 out).
    dim3 gg(D / 128, Mrows / 128);  // (8, 32)
    gemm_bf16_kernel<true><<<gg, 256, 0, stream>>>(xb, WqT, bq, Qb, Mrows, D, D);
    gemm_bf16_kernel<true><<<gg, 256, 0, stream>>>(xb, WkT, bk, Kb, Mrows, D, D);
    gemm_bf16_kernel<true><<<gg, 256, 0, stream>>>(xb, WvT, bv, Vb, Mrows, D, D);

    // 3) Attention (flash-style, WMMA for both matmuls).
    attn_kernel<<<dim3(32, 16, 2), 256, 0, stream>>>(Qb, Kb, Vb, Ab);

    // 4) Output projection -> f32 result.
    gemm_bf16_kernel<false><<<gg, 256, 0, stream>>>(Ab, WoT, bo, d_out, Mrows, D, D);
}
